// BipartiteGCNModel_56435870269558
// MI455X (gfx1250) — matrix operand order repaired and checked by reading, not compiled, not run on hardware
//
#include <hip/hip_runtime.h>
#include <hip/hip_bf16.h>

typedef __attribute__((ext_vector_type(16))) _Float16 v16h;
typedef __attribute__((ext_vector_type(8)))  _Float16 h8;
typedef __attribute__((ext_vector_type(8)))  float    v8f;
typedef __attribute__((ext_vector_type(4)))  unsigned int v4u;
typedef __attribute__((ext_vector_type(8)))  int      v8i;
typedef __attribute__((ext_vector_type(4)))  int      v4i;

#define EMB 64
#define WPB 4              // waves per block (wave32)
#define TPB (WPB * 32)

#define WMMA_F16(a, b, c) \
  __builtin_amdgcn_wmma_f32_16x16x32_f16(false, (a), false, (b), (short)0, (c), false, false)

// ---------------------------------------------------------------------------
// Fragment helpers. All data is kept FRAGMENT-MAJOR so every WMMA operand is
// a contiguous 32-byte per-lane read (2x b128), no per-element gathers.
//   A region  : [kt][lane][16 halves]            (per 16-row tile, in LDS)
//   B weights : [ (part*2+kt)*4 + nt ][lane][16] (pre-swizzled f16, global)
// Layout per CDNA5 ISA §7.12.2 (wave32).
// ---------------------------------------------------------------------------
__device__ __forceinline__ v16h ld_frag(const _Float16* p) {
  return *(const v16h*)p;
}

__device__ __forceinline__ h8 pack8(const float* v) {
  h8 p;
#pragma unroll
  for (int j = 0; j < 8; ++j) p[j] = (_Float16)v[j];
  return p;
}

// Stage one row (m) of a 16x64 fp32 tile (this lane holds cols hi*32..+31)
// into fragment-major A. Inverse of the A-fragment K striping:
//   cols c0+[0..7]  -> lane m    elems 0..7      cols c0+[8..15] -> lane m+16 elems 0..7
//   cols c0+[16..23]-> lane m    elems 8..15     cols c0+[24..31]-> lane m+16 elems 8..15
__device__ __forceinline__ void stage_a(const float* vals, _Float16* sAf, int m, int hi) {
  _Float16* b = sAf + (hi * 32 + m) * 16;     // kt == hi for this lane's columns
  *(h8*)(b)           = pack8(vals + 0);
  *(h8*)(b + 256)     = pack8(vals + 8);      // lane m+16
  *(h8*)(b + 8)       = pack8(vals + 16);
  *(h8*)(b + 256 + 8) = pack8(vals + 24);
}

// ---------------------------------------------------------------------------
// Weight pre-swizzle: f32 [K x 64] row-major -> f16 fragment-major in global
// workspace. One block. Stages the raw weights into LDS with the Tensor Data
// Mover (async DMA, TENSORcnt) when available.
// ---------------------------------------------------------------------------
__global__ __launch_bounds__(128) void wconvert_kernel(
    const float* __restrict__ W, int K, _Float16* __restrict__ Wf)
{
  __shared__ __align__(16) float sS[128 * EMB];   // up to 32KB (K<=128)

#if defined(__gfx1250__) && __has_builtin(__builtin_amdgcn_tensor_load_to_lds) && \
    __has_builtin(__builtin_amdgcn_s_wait_tensorcnt)
  if (threadIdx.x < 32) {                         // one wave issues the DMA
    const unsigned ne  = (unsigned)(K * EMB);     // elements (4B each)
    const unsigned lds = (unsigned)(unsigned long long)(size_t)&sS[0]; // LDS byte offset
    const unsigned long long ga = (unsigned long long)(size_t)W;
    v4u g0;
    g0[0] = 1u;                                          // count=1 (valid user D#)
    g0[1] = lds;                                         // lds_addr
    g0[2] = (unsigned)(ga & 0xFFFFFFFFu);                // global_addr[31:0]
    g0[3] = (unsigned)((ga >> 32) & 0x1FFFFFFu) | (2u << 30); // addr[56:32] | type=2
    v8i g1;
    g1[0] = (int)(2u << 16);                             // data_size = 4B
    g1[1] = (int)((ne & 0xFFFFu) << 16);                 // tensor_dim0 lo16
    g1[2] = (int)((ne >> 16) | (1u << 16));              // tensor_dim0 hi16 | tensor_dim1=1
    g1[3] = (int)((ne & 0xFFFFu) << 16);                 // tile_dim0 = ne
    g1[4] = 1;                                           // tile_dim1 = 1
    g1[5] = (int)ne;                                     // tensor_dim0_stride
    g1[6] = 0; g1[7] = 0;
    v4i g2 = {0, 0, 0, 0}, g3 = {0, 0, 0, 0};
    v8i g4 = {0, 0, 0, 0, 0, 0, 0, 0};
    __builtin_amdgcn_tensor_load_to_lds(g0, g1, g2, g3, g4, 0);
    __builtin_amdgcn_s_wait_tensorcnt(0);
  }
#else
  for (int i = threadIdx.x; i < K * EMB; i += 128) sS[i] = W[i];
#endif
  __syncthreads();

  // task = (n, 16-row k block): 16 strided LDS reads -> two b128 global stores
  const int ntask = (K >> 4) * EMB;
  for (int t = threadIdx.x; t < ntask; t += 128) {
    const int n    = t & 63;
    const int kb   = t >> 6;
    const int part = kb >> 2;
    const int kt   = (kb >> 1) & 1;
    const int hi   = kb & 1;
    const int lane = (n & 15) + 16 * hi;
    const int nt   = n >> 4;
    float v0[8], v1[8];
#pragma unroll
    for (int j = 0; j < 8; ++j) v0[j] = sS[(kb * 16 - 8 * hi + j) * EMB + n];
#pragma unroll
    for (int j = 0; j < 8; ++j) v1[j] = sS[(kb * 16 + 16 - 8 * hi + j) * EMB + n];
    _Float16* dst = Wf + (size_t)(((part * 2 + kt) * 4 + nt) * 32 + lane) * 16;
    *(h8*)dst       = pack8(v0);
    *(h8*)(dst + 8) = pack8(v1);
  }
}

// ---------------------------------------------------------------------------
// Y[Nx64] = op( X[Nx64] @ W[64x64] + bias ), W pre-swizzled f16 fragments.
// One wave per 16-row tile: 8x v_wmma_f32_16x16x32_f16.
// ---------------------------------------------------------------------------
__global__ __launch_bounds__(TPB) void gemm_n64_kernel(
    const float* __restrict__ X, const _Float16* __restrict__ Wf,
    const float* __restrict__ bias, float* __restrict__ Y,
    int N, int relu)
{
  __shared__ __align__(32) _Float16 sAf[WPB][2 * 32 * 16];

  const int lane = threadIdx.x & 31;
  const int w    = threadIdx.x >> 5;
  const int r0   = (blockIdx.x * WPB + w) * 16;
  const int m    = lane & 15;
  const int hi   = lane >> 4;
  const int c0   = hi * 32;
  const int row  = r0 + m;

  float vals[32];
  if (row < N) {
    const float4* xp = (const float4*)(X + (size_t)row * EMB + c0);
#pragma unroll
    for (int q = 0; q < 8; ++q) {
      const float4 v = xp[q];
      vals[q*4+0] = v.x; vals[q*4+1] = v.y; vals[q*4+2] = v.z; vals[q*4+3] = v.w;
    }
  } else {
#pragma unroll
    for (int q = 0; q < 32; ++q) vals[q] = 0.f;
  }
  stage_a(vals, sAf[w], m, hi);
  __syncthreads();

  v8f acc[4];
#pragma unroll
  for (int nt = 0; nt < 4; ++nt) {
    const float b0 = bias ? bias[nt * 16 + (lane & 15)] : 0.f;
#pragma unroll
    for (int j = 0; j < 8; ++j) acc[nt][j] = b0;
  }
#pragma unroll
  for (int kt = 0; kt < 2; ++kt) {
    const v16h a = ld_frag(sAf[w] + (kt * 32 + lane) * 16);
#pragma unroll
    for (int nt = 0; nt < 4; ++nt) {
      const v16h b = ld_frag(Wf + (size_t)((kt * 4 + nt) * 32 + lane) * 16);
      acc[nt] = WMMA_F16(a, b, acc[nt]);
    }
  }
#pragma unroll
  for (int nt = 0; nt < 4; ++nt) {
    const int n = nt * 16 + (lane & 15);
#pragma unroll
    for (int j = 0; j < 8; ++j) {
      const int rr = r0 + j + 8 * hi;
      float v = acc[nt][j];
      if (relu) v = fmaxf(v, 0.f);
      if (rr < N) Y[(size_t)rr * EMB + n] = v;
    }
  }
}

// ---------------------------------------------------------------------------
// First embed layer: h1 = ReLU( LN_d(x) @ W1[dx64] + b1 ), d = 5 or 19.
// ---------------------------------------------------------------------------
__global__ void embed_first_kernel(
    const float* __restrict__ X, int N, int D,
    const float* __restrict__ lng, const float* __restrict__ lnb,
    const float* __restrict__ W1, const float* __restrict__ b1,
    float* __restrict__ H1)
{
  const int gid  = blockIdx.x * blockDim.x + threadIdx.x;
  const int node = gid >> 6;
  const int n    = gid & 63;
  if (node >= N) return;

  float xs[20];
  float mu = 0.f;
  for (int d = 0; d < D; ++d) { xs[d] = X[(size_t)node * D + d]; mu += xs[d]; }
  mu /= (float)D;
  float var = 0.f;
  for (int d = 0; d < D; ++d) { const float t = xs[d] - mu; var += t * t; }
  var /= (float)D;
  const float rstd = rsqrtf(var + 1e-5f);

  float s = b1[n];
  for (int d = 0; d < D; ++d) {
    const float xn = (xs[d] - mu) * rstd * lng[d] + lnb[d];
    s += xn * W1[d * EMB + n];
  }
  H1[(size_t)node * EMB + n] = fmaxf(s, 0.f);
}

// ---------------------------------------------------------------------------
// Edge message kernel (16 edges per wave tile):
//   m = ReLU(LN( Rl[dst] + Rr[src] + (edge_ln_b*fe_W + fl_b) ))
//   agg[dst] += m @ fin_W + fin_b       (8x WMMA + f32 atomics)
// ---------------------------------------------------------------------------
__global__ __launch_bounds__(TPB) void edge_msg_kernel(
    const int* __restrict__ srcIdx, const int* __restrict__ dstIdx, int E,
    const float* __restrict__ Rl, const float* __restrict__ Rr,
    const float* __restrict__ feW, const float* __restrict__ flb,
    const float* __restrict__ elnb,
    const float* __restrict__ gin, const float* __restrict__ bin,
    const _Float16* __restrict__ finWf, const float* __restrict__ finb,
    float* __restrict__ agg)
{
  __shared__ __align__(32) _Float16 sAf[WPB][2 * 32 * 16];
  __shared__ int sDst[WPB][16];

  const int lane = threadIdx.x & 31;
  const int w    = threadIdx.x >> 5;
  const int e0   = (blockIdx.x * WPB + w) * 16;
  const int m    = lane & 15;
  const int hi   = lane >> 4;
  const int c0   = hi * 32;
  const int e    = e0 + m;
  const bool valid = (e < E);

  const int d = valid ? dstIdx[e] : 0;
  const int s = valid ? srcIdx[e] : 0;
  if (hi == 0) sDst[w][m] = d;

  const float eb = elnb[0];
  const float4* lp = (const float4*)(Rl + (size_t)d * EMB + c0);
  const float4* rp = (const float4*)(Rr + (size_t)s * EMB + c0);

  float vals[32];
  float sum = 0.f, ss = 0.f;
#pragma unroll
  for (int q = 0; q < 8; ++q) {
    const float4 a4 = lp[q];
    const float4 b4 = rp[q];
#pragma unroll
    for (int ii = 0; ii < 4; ++ii) {
      const int c = c0 + q * 4 + ii;
      const float xi = (ii == 0 ? a4.x : ii == 1 ? a4.y : ii == 2 ? a4.z : a4.w)
                     + (ii == 0 ? b4.x : ii == 1 ? b4.y : ii == 2 ? b4.z : b4.w)
                     + eb * feW[c] + flb[c];
      vals[q * 4 + ii] = xi;
      sum += xi; ss += xi * xi;
    }
  }
  sum += __shfl_xor(sum, 16, 32);   // lane pair (L, L^16) = two halves of row m
  ss  += __shfl_xor(ss, 16, 32);
  const float mean = sum * (1.f / 64.f);
  const float var  = fmaxf(ss * (1.f / 64.f) - mean * mean, 0.f);
  const float rstd = rsqrtf(var + 1e-5f);
#pragma unroll
  for (int q = 0; q < 32; ++q) {
    const int c = c0 + q;
    vals[q] = fmaxf((vals[q] - mean) * rstd * gin[c] + bin[c], 0.f);
  }
  stage_a(vals, sAf[w], m, hi);
  __syncthreads();

  v8f acc[4];
#pragma unroll
  for (int nt = 0; nt < 4; ++nt) {
    const float b0 = finb[nt * 16 + (lane & 15)];
#pragma unroll
    for (int j = 0; j < 8; ++j) acc[nt][j] = b0;
  }
#pragma unroll
  for (int kt = 0; kt < 2; ++kt) {
    const v16h a = ld_frag(sAf[w] + (kt * 32 + lane) * 16);
#pragma unroll
    for (int nt = 0; nt < 4; ++nt) {
      const v16h b = ld_frag(finWf + (size_t)((kt * 4 + nt) * 32 + lane) * 16);
      acc[nt] = WMMA_F16(a, b, acc[nt]);
    }
  }
  // scatter-add (segment_sum over dst)
#pragma unroll
  for (int nt = 0; nt < 4; ++nt) {
    const int n = nt * 16 + (lane & 15);
#pragma unroll
    for (int j = 0; j < 8; ++j) {
      const int mm = j + 8 * hi;
      if (e0 + mm < E) {
        const int dd = sDst[w][mm];
        unsafeAtomicAdd(&agg[(size_t)dd * EMB + n], acc[nt][j]);
      }
    }
  }
}

// ---------------------------------------------------------------------------
// Post kernel: H = ReLU( [ LN(agg) , right ] @ W1[128x64] + b1 )  (K=128)
// ---------------------------------------------------------------------------
__global__ __launch_bounds__(TPB) void post_concat_kernel(
    const float* __restrict__ agg, const float* __restrict__ right,
    const float* __restrict__ png, const float* __restrict__ pnb,
    const _Float16* __restrict__ W1f, const float* __restrict__ b1,
    float* __restrict__ H, int N)
{
  __shared__ __align__(32) _Float16 sAf[WPB][2][2 * 32 * 16];

  const int lane = threadIdx.x & 31;
  const int w    = threadIdx.x >> 5;
  const int r0   = (blockIdx.x * WPB + w) * 16;
  const int m    = lane & 15;
  const int hi   = lane >> 4;
  const int c0   = hi * 32;
  const int row  = r0 + m;

  float vals[32];
  if (row < N) {
    // part 0: LN(agg row)
    const float4* ap = (const float4*)(agg + (size_t)row * EMB + c0);
    float sum = 0.f, ss = 0.f;
#pragma unroll
    for (int q = 0; q < 8; ++q) {
      const float4 v = ap[q];
      vals[q*4+0] = v.x; vals[q*4+1] = v.y; vals[q*4+2] = v.z; vals[q*4+3] = v.w;
      sum += v.x + v.y + v.z + v.w;
      ss  += v.x*v.x + v.y*v.y + v.z*v.z + v.w*v.w;
    }
    sum += __shfl_xor(sum, 16, 32);
    ss  += __shfl_xor(ss, 16, 32);
    const float mean = sum * (1.f / 64.f);
    const float var  = fmaxf(ss * (1.f / 64.f) - mean * mean, 0.f);
    const float rstd = rsqrtf(var + 1e-5f);
#pragma unroll
    for (int q = 0; q < 32; ++q) {
      const int c = c0 + q;
      vals[q] = (vals[q] - mean) * rstd * png[c] + pnb[c];
    }
    stage_a(vals, sAf[w][0], m, hi);
    // part 1: right row
    const float4* rp = (const float4*)(right + (size_t)row * EMB + c0);
#pragma unroll
    for (int q = 0; q < 8; ++q) {
      const float4 v = rp[q];
      vals[q*4+0] = v.x; vals[q*4+1] = v.y; vals[q*4+2] = v.z; vals[q*4+3] = v.w;
    }
    stage_a(vals, sAf[w][1], m, hi);
  } else {
#pragma unroll
    for (int q = 0; q < 32; ++q) vals[q] = 0.f;
    stage_a(vals, sAf[w][0], m, hi);
    stage_a(vals, sAf[w][1], m, hi);
  }
  __syncthreads();

  v8f acc[4];
#pragma unroll
  for (int nt = 0; nt < 4; ++nt) {
    const float b0 = b1[nt * 16 + (lane & 15)];
#pragma unroll
    for (int j = 0; j < 8; ++j) acc[nt][j] = b0;
  }
#pragma unroll
  for (int p = 0; p < 2; ++p) {
#pragma unroll
    for (int kt = 0; kt < 2; ++kt) {
      const v16h a = ld_frag(sAf[w][p] + (kt * 32 + lane) * 16);
#pragma unroll
      for (int nt = 0; nt < 4; ++nt) {
        const v16h b = ld_frag(W1f + (size_t)(((p * 2 + kt) * 4 + nt) * 32 + lane) * 16);
        acc[nt] = WMMA_F16(a, b, acc[nt]);
      }
    }
  }
#pragma unroll
  for (int nt = 0; nt < 4; ++nt) {
    const int n = nt * 16 + (lane & 15);
#pragma unroll
    for (int j = 0; j < 8; ++j) {
      const int rr = r0 + j + 8 * hi;
      if (rr < N) H[(size_t)rr * EMB + n] = fmaxf(acc[nt][j], 0.f);
    }
  }
}

// ---------------------------------------------------------------------------
// Head: out[i] = sum_n H[i][n] * w2[n]
// ---------------------------------------------------------------------------
__global__ void head_dot_kernel(const float* __restrict__ H, const float* __restrict__ w2,
                                float* __restrict__ out, int N)
{
  const int i = blockIdx.x * blockDim.x + threadIdx.x;
  if (i >= N) return;
  const float4* hp = (const float4*)(H + (size_t)i * EMB);
  float s = 0.f;
#pragma unroll
  for (int q = 0; q < 16; ++q) {
    const float4 h4 = hp[q];
    s += h4.x * w2[q*4+0] + h4.y * w2[q*4+1] + h4.z * w2[q*4+2] + h4.w * w2[q*4+3];
  }
  out[i] = s;
}

// ---------------------------------------------------------------------------
// Host-side orchestration
// ---------------------------------------------------------------------------
enum {
  P_CONS_LN_G = 0, P_CONS_LN_B, P_CONS_W1, P_CONS_B1, P_CONS_W2, P_CONS_B2,
  P_VAR_LN_G,      P_VAR_LN_B,  P_VAR_W1,  P_VAR_B1,  P_VAR_W2,  P_VAR_B2,
  P_EDGE_LN_G,     P_EDGE_LN_B,
  P_VC = 14, P_CV = 28,
  P_HEAD_W1 = 42, P_HEAD_B1 = 43, P_HEAD_W2 = 44,
  NPARAMS = 45
};
enum { C_FL_W = 0, C_FL_B, C_FE_W, C_FR_W, C_FIN_LN_G, C_FIN_LN_B,
       C_FIN_W, C_FIN_B, C_POST_LN_G, C_POST_LN_B,
       C_OUT_W1, C_OUT_B1, C_OUT_W2, C_OUT_B2 };

extern "C" void kernel_launch(void* const* d_in, const int* in_sizes, int n_in,
                              void* d_out, int out_size, void* d_ws, size_t ws_size,
                              hipStream_t stream)
{
  const float* cons_x = (const float*)d_in[0];
  const int*   eidx   = (const int*)d_in[1];
  const float* var_x  = (const float*)d_in[3];
  // d_in[2] (edge_features) is provably dead: LN over a size-1 axis == beta.

  const float* P[NPARAMS];
  for (int i = 0; i < NPARAMS && (4 + i) < n_in; ++i) P[i] = (const float*)d_in[4 + i];

  const int NC = in_sizes[0] / 5;
  const int NV = in_sizes[3] / 19;
  const int E  = in_sizes[1] / 2;
  const int NMAX = (NC > NV) ? NC : NV;

  // ---- workspace layout ----
  float* base  = (float*)d_ws;
  const size_t per = (size_t)NMAX * EMB;
  float* c_emb = base;
  float* v_emb = c_emb + (size_t)NC * EMB;
  float* c_new = v_emb + (size_t)NV * EMB;
  float* v_new = c_new + (size_t)NC * EMB;
  float* Rl    = v_new + (size_t)NV * EMB;
  float* Rr    = Rl + per;
  float* agg   = Rr + per;
  float* tmp   = agg + per;
  _Float16* wf = (_Float16*)(tmp + per);   // fragment-major f16 weights
  const size_t W64 = 64 * EMB;             // 4096 halves
  const size_t W128 = 128 * EMB;           // 8192 halves
  _Float16* wf_consW2 = wf;
  _Float16* wf_varW2  = wf_consW2 + W64;
  _Float16* wf_vc_fl  = wf_varW2 + W64;
  _Float16* wf_vc_fr  = wf_vc_fl + W64;
  _Float16* wf_vc_fin = wf_vc_fr + W64;
  _Float16* wf_vc_o1  = wf_vc_fin + W64;
  _Float16* wf_vc_o2  = wf_vc_o1 + W128;
  _Float16* wf_cv_fl  = wf_vc_o2 + W64;
  _Float16* wf_cv_fr  = wf_cv_fl + W64;
  _Float16* wf_cv_fin = wf_cv_fr + W64;
  _Float16* wf_cv_o1  = wf_cv_fin + W64;
  _Float16* wf_cv_o2  = wf_cv_o1 + W128;
  _Float16* wf_head1  = wf_cv_o2 + W64;
  (void)ws_size;

  const int cons_off = 0;   // edge_indices row 0 = constraint side
  const int var_off  = E;   // edge_indices row 1 = variable side

  auto tileGrid = [](int N) { return (((N + 15) / 16) + WPB - 1) / WPB; };
  const int gC  = tileGrid(NC);
  const int gV  = tileGrid(NV);
  const int gE  = tileGrid(E);
  const int gVt = (NV * EMB + 255) / 256;
  const int gCt = (NC * EMB + 255) / 256;

  // ---- weight pre-swizzle (TDM -> LDS -> fragment-major f16) ----
  wconvert_kernel<<<1, 128, 0, stream>>>(P[P_CONS_W2],          64,  wf_consW2);
  wconvert_kernel<<<1, 128, 0, stream>>>(P[P_VAR_W2],           64,  wf_varW2);
  wconvert_kernel<<<1, 128, 0, stream>>>(P[P_VC + C_FL_W],      64,  wf_vc_fl);
  wconvert_kernel<<<1, 128, 0, stream>>>(P[P_VC + C_FR_W],      64,  wf_vc_fr);
  wconvert_kernel<<<1, 128, 0, stream>>>(P[P_VC + C_FIN_W],     64,  wf_vc_fin);
  wconvert_kernel<<<1, 128, 0, stream>>>(P[P_VC + C_OUT_W1],    128, wf_vc_o1);
  wconvert_kernel<<<1, 128, 0, stream>>>(P[P_VC + C_OUT_W2],    64,  wf_vc_o2);
  wconvert_kernel<<<1, 128, 0, stream>>>(P[P_CV + C_FL_W],      64,  wf_cv_fl);
  wconvert_kernel<<<1, 128, 0, stream>>>(P[P_CV + C_FR_W],      64,  wf_cv_fr);
  wconvert_kernel<<<1, 128, 0, stream>>>(P[P_CV + C_FIN_W],     64,  wf_cv_fin);
  wconvert_kernel<<<1, 128, 0, stream>>>(P[P_CV + C_OUT_W1],    128, wf_cv_o1);
  wconvert_kernel<<<1, 128, 0, stream>>>(P[P_CV + C_OUT_W2],    64,  wf_cv_o2);
  wconvert_kernel<<<1, 128, 0, stream>>>(P[P_HEAD_W1],          64,  wf_head1);

  // ---- node embeddings ----
  embed_first_kernel<<<gCt, 256, 0, stream>>>(cons_x, NC, 5,
      P[P_CONS_LN_G], P[P_CONS_LN_B], P[P_CONS_W1], P[P_CONS_B1], tmp);
  gemm_n64_kernel<<<gC, TPB, 0, stream>>>(tmp, wf_consW2, P[P_CONS_B2], c_emb, NC, 1);

  embed_first_kernel<<<gVt, 256, 0, stream>>>(var_x, NV, 19,
      P[P_VAR_LN_G], P[P_VAR_LN_B], P[P_VAR_W1], P[P_VAR_B1], tmp);
  gemm_n64_kernel<<<gV, TPB, 0, stream>>>(tmp, wf_varW2, P[P_VAR_B2], v_emb, NV, 1);

  // ---- vc pass: left = v_emb (src=var), right = c_emb (dst=cons) -> c_new ----
  {
    const float* const* Q = P + P_VC;
    gemm_n64_kernel<<<gC, TPB, 0, stream>>>(c_emb, wf_vc_fl, nullptr, Rl, NC, 0);
    gemm_n64_kernel<<<gV, TPB, 0, stream>>>(v_emb, wf_vc_fr, nullptr, Rr, NV, 0);
    (void)hipMemsetAsync(agg, 0, (size_t)NC * EMB * sizeof(float), stream);
    edge_msg_kernel<<<gE, TPB, 0, stream>>>(
        eidx + var_off, eidx + cons_off, E, Rl, Rr,
        Q[C_FE_W], Q[C_FL_B], P[P_EDGE_LN_B],
        Q[C_FIN_LN_G], Q[C_FIN_LN_B], wf_vc_fin, Q[C_FIN_B], agg);
    post_concat_kernel<<<gC, TPB, 0, stream>>>(
        agg, c_emb, Q[C_POST_LN_G], Q[C_POST_LN_B], wf_vc_o1, Q[C_OUT_B1], tmp, NC);
    gemm_n64_kernel<<<gC, TPB, 0, stream>>>(tmp, wf_vc_o2, Q[C_OUT_B2], c_new, NC, 0);
  }

  // ---- cv pass: left = c_new (src=cons), right = v_emb (dst=var) -> v_new ----
  {
    const float* const* Q = P + P_CV;
    gemm_n64_kernel<<<gV, TPB, 0, stream>>>(v_emb, wf_cv_fl, nullptr, Rl, NV, 0);
    gemm_n64_kernel<<<gC, TPB, 0, stream>>>(c_new, wf_cv_fr, nullptr, Rr, NC, 0);
    (void)hipMemsetAsync(agg, 0, (size_t)NV * EMB * sizeof(float), stream);
    edge_msg_kernel<<<gE, TPB, 0, stream>>>(
        eidx + cons_off, eidx + var_off, E, Rl, Rr,
        Q[C_FE_W], Q[C_FL_B], P[P_EDGE_LN_B],
        Q[C_FIN_LN_G], Q[C_FIN_LN_B], wf_cv_fin, Q[C_FIN_B], agg);
    post_concat_kernel<<<gV, TPB, 0, stream>>>(
        agg, v_emb, Q[C_POST_LN_G], Q[C_POST_LN_B], wf_cv_o1, Q[C_OUT_B1], tmp, NV);
    gemm_n64_kernel<<<gV, TPB, 0, stream>>>(tmp, wf_cv_o2, Q[C_OUT_B2], v_new, NV, 0);
  }

  // ---- head ----
  gemm_n64_kernel<<<gV, TPB, 0, stream>>>(v_new, wf_head1, P[P_HEAD_B1], tmp, NV, 1);
  head_dot_kernel<<<(NV + 255) / 256, 256, 0, stream>>>(tmp, P[P_HEAD_W2], (float*)d_out, NV);
  (void)out_size;
}
